// hGRU_14078902796397
// MI455X (gfx1250) — compile-verified, matrix operand order checked by reading, not run on hardware
//
#include <hip/hip_runtime.h>
#include <hip/hip_bf16.h>
#include <cstdint>

// ---------------- problem constants ----------------
#define BB 16
#define CC 25
#define HH 150
#define WW 150
#define PXC (HH*WW)          // 22500 pixels per image
#define NPC (BB*PXC)         // 360000 per-channel elements (BN population)
#define NEL (BB*CC*PXC)      // 9,000,000 elements in a [B,C,H,W] tensor
#define HP 164               // padded height (150 + 2*7)
#define WPAD 176             // padded width (room for tile overshoot)
#define CPAD 32              // channels padded 25 -> 32 for WMMA K
#define XTILES 10            // ceil(150/16)
#define YPAIRS 75            // 150/2 row pairs per image
#define NTAP 225             // 15*15
#define WPACK_ELEMS (NTAP*2*32*16)   // packed weight halfs per conv

typedef __attribute__((ext_vector_type(16))) _Float16 v16h;
typedef __attribute__((ext_vector_type(8)))  _Float16 v8h;
typedef __attribute__((ext_vector_type(8)))  float    v8f;

__device__ __forceinline__ float softplusf(float v) {
  return v > 20.f ? v : log1pf(expf(v));
}
__device__ __forceinline__ float sigmoidf(float v) {
  return 1.f / (1.f + expf(-v));
}

// ---------------- weight packing into WMMA B-register layout ----------------
// wp layout: [tap(225)][ntile(2)][lane(32)][elem(16)] halfs.
// lane n%16 = output channel within tile; K index per elem mirrors the f16
// 16x32 A/B VGPR striping: k = (e&7) + 8*((e>>3)*2 + (lane>>4)).
__global__ void pack_w_kernel(const float* __restrict__ w, _Float16* __restrict__ wp) {
  int idx = blockIdx.x * blockDim.x + threadIdx.x;
  if (idx >= WPACK_ELEMS) return;
  int e    = idx & 15;
  int lane = (idx >> 4) & 31;
  int nt   = (idx >> 9) & 1;
  int tap  = idx >> 10;
  int ky = tap / 15, kx = tap % 15;
  int co = nt * 16 + (lane & 15);
  int k  = (e & 7) + 8 * (((e >> 3) << 1) + (lane >> 4));
  float v = 0.f;
  if (co < CC && k < CC)
    v = w[((co * CC + k) * 15 + ky) * 15 + kx];   // OIHW
  wp[idx] = (_Float16)v;
}

// ---------------- conv0 (7x7, 4->25, pad 3) then square ----------------
__global__ void conv0_kernel(const float* __restrict__ x, const float* __restrict__ w,
                             const float* __restrict__ b, float* __restrict__ xin) {
  int i = blockIdx.x * blockDim.x + threadIdx.x;
  if (i >= NEL) return;
  int xx = i % WW;
  int yy = (i / WW) % HH;
  int c  = (i / PXC) % CC;
  int bb = i / (CC * PXC);
  float s = b[c];
  for (int ci = 0; ci < 4; ++ci) {
    for (int ky = 0; ky < 7; ++ky) {
      int iy = yy + ky - 3;
      if (iy < 0 || iy >= HH) continue;
      for (int kx = 0; kx < 7; ++kx) {
        int ix = xx + kx - 3;
        if (ix < 0 || ix >= WW) continue;
        s += x[((bb * 4 + ci) * HH + iy) * WW + ix] *
             w[((c * 4 + ci) * 7 + ky) * 7 + kx];
      }
    }
  }
  xin[i] = s * s;
}

// ---------------- g1 = sigmoid(u1 * h + b1); write h*g1 into padded f16 ----------------
__global__ void g1hg_kernel(const float* __restrict__ h,
                            const float* __restrict__ u1w, const float* __restrict__ u1b,
                            _Float16* __restrict__ pad) {
  int p = blockIdx.x * blockDim.x + threadIdx.x;
  if (p >= NPC) return;
  int xx = p % WW;
  int yy = (p / WW) % HH;
  int bb = p / PXC;
  float hv[CC];
  size_t base = (size_t)bb * CC * PXC + (size_t)yy * WW + xx;
  for (int c = 0; c < CC; ++c) hv[c] = h[base + (size_t)c * PXC];
  _Float16* op = pad + (((size_t)(bb * HP + yy + 7) * WPAD) + xx + 7) * CPAD;
  for (int c = 0; c < CC; ++c) {
    float s = u1b[c];
    for (int ci = 0; ci < CC; ++ci) s += u1w[c * CC + ci] * hv[ci];
    op[c] = (_Float16)(hv[c] * sigmoidf(s));
  }
}

// ---------------- 15x15 conv as implicit GEMM via WMMA f16 ----------------
// One wave computes TWO adjacent output rows of a 16(pixel-x) x 32(out-chan)
// tile, sharing every B tile across both rows (1.5 VMEM per WMMA).
__global__ __launch_bounds__(128) void conv15_wmma_kernel(
    const _Float16* __restrict__ pad,   // [B][HP][WPAD][CPAD] f16, halo zero
    const _Float16* __restrict__ wp,    // packed [225][2][32][16]
    float* __restrict__ out)            // [B][25][150][150] f32
{
  const int lane = threadIdx.x & 31;
  const int wave = threadIdx.x >> 5;
  const int w = blockIdx.x * 4 + wave;
  if (w >= BB * YPAIRS * XTILES) return;    // wave-uniform
  const int xt = w % XTILES;
  const int yp = (w / XTILES) % YPAIRS;
  const int bb = w / (YPAIRS * XTILES);
  const int yo = yp * 2;                    // rows yo, yo+1
  const int x0 = xt * 16;
  const int m   = lane & 15;                // pixel within tile (M)
  const int sel = lane >> 4;                // K-half selector

  v8f accA0 = {};   // row yo,   co 0..15
  v8f accA1 = {};   // row yo,   co 16..31
  v8f accB0 = {};   // row yo+1, co 0..15
  v8f accB1 = {};   // row yo+1, co 16..31

  for (int ky = 0; ky < 15; ++ky) {
    const _Float16* rowp0 = pad +
        ((size_t)(bb * HP + yo + ky) * WPAD + (size_t)(x0 + m)) * CPAD + sel * 8;
    const _Float16* rowp1 = rowp0 + (size_t)WPAD * CPAD;
    const _Float16* wrow = wp + (size_t)(ky * 15) * (2 * 32 * 16);
    for (int kx = 0; kx < 15; ++kx) {
      // A tiles: 16 pixels x 32 channels (f16), per-lane two b128 loads each
      const _Float16* ap0 = rowp0 + kx * CPAD;
      const _Float16* ap1 = rowp1 + kx * CPAD;
      union { v16h v; v8h h[2]; } ua0, ua1;
      ua0.h[0] = *(const v8h*)(ap0);        // K 0..7 | 8..15
      ua0.h[1] = *(const v8h*)(ap0 + 16);   // K 16..23 | 24..31
      ua1.h[0] = *(const v8h*)(ap1);
      ua1.h[1] = *(const v8h*)(ap1 + 16);
      // B tiles: pre-packed per-lane register image, shared by both rows
      const _Float16* bp = wrow + kx * (2 * 32 * 16) + lane * 16;
      v16h b0 = *(const v16h*)(bp);
      v16h b1 = *(const v16h*)(bp + 32 * 16);
      accA0 = __builtin_amdgcn_wmma_f32_16x16x32_f16(
                  false, ua0.v, false, b0, (short)0, accA0, false, false);
      accA1 = __builtin_amdgcn_wmma_f32_16x16x32_f16(
                  false, ua0.v, false, b1, (short)0, accA1, false, false);
      accB0 = __builtin_amdgcn_wmma_f32_16x16x32_f16(
                  false, ua1.v, false, b0, (short)0, accB0, false, false);
      accB1 = __builtin_amdgcn_wmma_f32_16x16x32_f16(
                  false, ua1.v, false, b1, (short)0, accB1, false, false);
    }
  }

  // D layout: VGPR r, lanes 0-15: M=r, N=lane; lanes 16-31: M=r+8, N=lane-16
  const int n = lane & 15;
  for (int r = 0; r < 8; ++r) {
    int xx = x0 + r + 8 * sel;
    if (xx >= WW) continue;
    if (n < CC) {
      out[((size_t)(bb * CC + n) * HH + yo) * WW + xx]       = accA0[r];
      out[((size_t)(bb * CC + n) * HH + yo + 1) * WW + xx]   = accB0[r];
    }
    int co1 = n + 16;
    if (co1 < CC) {
      out[((size_t)(bb * CC + co1) * HH + yo) * WW + xx]     = accA1[r];
      out[((size_t)(bb * CC + co1) * HH + yo + 1) * WW + xx] = accB1[r];
    }
  }
}

// ---------------- BN statistics: deterministic two-stage reduction ----------------
__global__ void bnstats_kernel(const float* __restrict__ v,
                               float* __restrict__ part /* [25][64][2] */) {
  const int c = blockIdx.y;
  const int chunk = blockIdx.x;       // 64 chunks
  const int tid = threadIdx.x;
  float s = 0.f, s2 = 0.f;
  for (int i = chunk * 256 + tid; i < NPC; i += 64 * 256) {
    int bb = i / PXC, p = i % PXC;
    float val = v[(size_t)(bb * CC + c) * PXC + p];
    s += val; s2 += val * val;
  }
  __shared__ float sh[256], sh2[256];
  sh[tid] = s; sh2[tid] = s2;
  __syncthreads();
  for (int o = 128; o > 0; o >>= 1) {
    if (tid < o) { sh[tid] += sh[tid + o]; sh2[tid] += sh2[tid + o]; }
    __syncthreads();
  }
  if (tid == 0) {
    part[(c * 64 + chunk) * 2]     = sh[0];
    part[(c * 64 + chunk) * 2 + 1] = sh2[0];
  }
}

__global__ void bnfinal_kernel(const float* __restrict__ part,
                               float* __restrict__ stats /* [25][2] */) {
  int c = threadIdx.x;
  if (c >= CC) return;
  float s = 0.f, s2 = 0.f;
  for (int k = 0; k < 64; ++k) {
    s  += part[(c * 64 + k) * 2];
    s2 += part[(c * 64 + k) * 2 + 1];
  }
  stats[c * 2] = s;
  stats[c * 2 + 1] = s2;
}

// ---------------- ns1 = sp(xin - sp(BN(c1) * (alpha*h + mu))) -> padded f16 ----------------
__global__ void ns1_kernel(const float* __restrict__ craw, const float* __restrict__ stats,
                           const float* __restrict__ bnw, const float* __restrict__ bnb,
                           const float* __restrict__ xin, const float* __restrict__ h,
                           const float* __restrict__ alpha, const float* __restrict__ mu,
                           _Float16* __restrict__ pad) {
  int i = blockIdx.x * blockDim.x + threadIdx.x;
  if (i >= NEL) return;
  int p = i % PXC;
  int c = (i / PXC) % CC;
  int bb = i / (CC * PXC);
  float mean = stats[c * 2] * (1.f / (float)NPC);
  float var  = stats[c * 2 + 1] * (1.f / (float)NPC) - mean * mean;
  float c1 = (craw[i] - mean) * rsqrtf(var + 1e-3f) * bnw[c] + bnb[c];
  float inner = softplusf(c1 * (alpha[c] * h[i] + mu[c]));
  float ns = softplusf(xin[i] - inner);
  int xx = p % WW, yy = p / WW;
  pad[(((size_t)(bb * HP + yy + 7) * WPAD) + xx + 7) * CPAD + c] = (_Float16)ns;
}

// ---------------- h-update: g2 gate + mix + softplus ----------------
__global__ void hupd_kernel(const _Float16* __restrict__ pad,  // ns1 (padded f16)
                            const float* __restrict__ craw,    // c2 raw
                            const float* __restrict__ stats,
                            const float* __restrict__ bnw, const float* __restrict__ bnb,
                            const float* __restrict__ u2w, const float* __restrict__ u2b,
                            const float* __restrict__ kappa, const float* __restrict__ gam,
                            const float* __restrict__ wmix,
                            float* __restrict__ h) {
  int p = blockIdx.x * blockDim.x + threadIdx.x;
  if (p >= NPC) return;
  int xx = p % WW;
  int yy = (p / WW) % HH;
  int bb = p / PXC;
  const _Float16* np = pad + (((size_t)(bb * HP + yy + 7) * WPAD) + xx + 7) * CPAD;
  float ns[CC];
  for (int c = 0; c < CC; ++c) ns[c] = (float)np[c];
  size_t base = (size_t)bb * CC * PXC + (size_t)yy * WW + xx;
  for (int c = 0; c < CC; ++c) {
    float g = u2b[c];
    for (int ci = 0; ci < CC; ++ci) g += u2w[c * CC + ci] * ns[ci];
    float g2 = sigmoidf(g);
    float mean = stats[c * 2] * (1.f / (float)NPC);
    float var  = stats[c * 2 + 1] * (1.f / (float)NPC) - mean * mean;
    size_t idx = base + (size_t)c * PXC;
    float c2 = (craw[idx] - mean) * rsqrtf(var + 1e-3f) * bnw[c] + bnb[c];
    float h2 = softplusf(kappa[c] * ns[c] + gam[c] * c2 + wmix[c] * ns[c] * c2);
    float hold = h[idx];
    h[idx] = softplusf((1.f - g2) * hold + g2 * h2);
  }
}

// ---------------- readout: per-(b) partial sums of o and o^2 ----------------
__global__ void readout_reduce_kernel(const float* __restrict__ h,
                                      const float* __restrict__ cw,   // [2][25]
                                      const float* __restrict__ cb,   // [2]
                                      float* __restrict__ rp /* [16][64][4] */) {
  const int bb = blockIdx.y;
  const int chunk = blockIdx.x;  // 64
  const int tid = threadIdx.x;
  float s0 = 0.f, s1 = 0.f, q0 = 0.f, q1 = 0.f;
  for (int p = chunk * 256 + tid; p < PXC; p += 64 * 256) {
    float o0 = cb[0], o1 = cb[1];
    for (int c = 0; c < CC; ++c) {
      float hv = h[(size_t)(bb * CC + c) * PXC + p];
      o0 += cw[c] * hv;
      o1 += cw[CC + c] * hv;
    }
    s0 += o0; s1 += o1; q0 += o0 * o0; q1 += o1 * o1;
  }
  __shared__ float sh[4][256];
  sh[0][tid] = s0; sh[1][tid] = s1; sh[2][tid] = q0; sh[3][tid] = q1;
  __syncthreads();
  for (int o = 128; o > 0; o >>= 1) {
    if (tid < o)
      for (int k = 0; k < 4; ++k) sh[k][tid] += sh[k][tid + o];
    __syncthreads();
  }
  if (tid == 0)
    for (int k = 0; k < 4; ++k)
      rp[((size_t)bb * 64 + chunk) * 4 + k] = sh[k][0];
}

__global__ void readout_final_kernel(const float* __restrict__ rp,
                                     const float* __restrict__ bnow, const float* __restrict__ bnob,
                                     const float* __restrict__ fcw, const float* __restrict__ fcb,
                                     float* __restrict__ out) {
  if (threadIdx.x != 0 || blockIdx.x != 0) return;
  float S[BB][2], SQ[2] = {0.f, 0.f};
  for (int bb = 0; bb < BB; ++bb) {
    for (int c = 0; c < 2; ++c) {
      float s = 0.f;
      for (int k = 0; k < 64; ++k) s += rp[((size_t)bb * 64 + k) * 4 + c];
      S[bb][c] = s;
      float q = 0.f;
      for (int k = 0; k < 64; ++k) q += rp[((size_t)bb * 64 + k) * 4 + 2 + c];
      SQ[c] += q;
    }
  }
  const float invHW = 1.f / (float)PXC;
  const float invN  = 1.f / (float)NPC;
  float mc[2], rs[2];
  for (int c = 0; c < 2; ++c) {
    float sum = 0.f;
    for (int bb = 0; bb < BB; ++bb) sum += S[bb][c];
    float m = sum * invN;
    float var = SQ[c] * invN - m * m;
    mc[c] = m;
    rs[c] = rsqrtf(var + 1e-3f);
  }
  for (int bb = 0; bb < BB; ++bb) {
    float pb[2];
    for (int c = 0; c < 2; ++c)
      pb[c] = (S[bb][c] * invHW - mc[c]) * rs[c] * bnow[c] + bnob[c];
    for (int j = 0; j < 2; ++j)
      out[bb * 2 + j] = fcw[j * 2 + 0] * pb[0] + fcw[j * 2 + 1] * pb[1] + fcb[j];
  }
}

// ---------------- host-side orchestration ----------------
extern "C" void kernel_launch(void* const* d_in, const int* in_sizes, int n_in,
                              void* d_out, int out_size, void* d_ws, size_t ws_size,
                              hipStream_t stream) {
  (void)in_sizes; (void)n_in; (void)out_size; (void)ws_size;
  const float* x       = (const float*)d_in[0];
  const float* h0      = (const float*)d_in[1];
  const float* conv0_w = (const float*)d_in[2];
  const float* conv0_b = (const float*)d_in[3];
  const float* u1_w    = (const float*)d_in[4];
  const float* u1_b    = (const float*)d_in[5];
  const float* u2_w    = (const float*)d_in[6];
  const float* w_inh   = (const float*)d_in[7];
  const float* w_exc   = (const float*)d_in[8];
  const float* alpha   = (const float*)d_in[9];
  const float* mu      = (const float*)d_in[10];
  const float* gam     = (const float*)d_in[11];
  const float* kappa   = (const float*)d_in[12];
  const float* w_mix   = (const float*)d_in[13];
  const float* bn1_w   = (const float*)d_in[14];
  const float* bn1_b   = (const float*)d_in[15];
  const float* bn3_w   = (const float*)d_in[16];
  const float* bn3_b   = (const float*)d_in[17];
  const float* conv6_w = (const float*)d_in[18];
  const float* conv6_b = (const float*)d_in[19];
  const float* bno_w   = (const float*)d_in[20];
  const float* bno_b   = (const float*)d_in[21];
  const float* fc_w    = (const float*)d_in[22];
  const float* fc_b    = (const float*)d_in[23];
  // d_in[24]=epoch, d_in[25]=itr (unused); u2_b appended last
  const float* u2_b    = (const float*)d_in[26];

  uint8_t* ws = (uint8_t*)d_ws;
  size_t off = 0;
  auto carve = [&](size_t bytes) -> void* {
    void* p = ws + off;
    off += (bytes + 255) & ~(size_t)255;
    return p;
  };
  const size_t padBytes = sizeof(_Float16) * (size_t)BB * HP * WPAD * CPAD;
  float*    h     = (float*)carve(sizeof(float) * NEL);
  float*    xin   = (float*)carve(sizeof(float) * NEL);
  float*    craw  = (float*)carve(sizeof(float) * NEL);
  _Float16* pad   = (_Float16*)carve(padBytes);
  _Float16* wpI   = (_Float16*)carve(sizeof(_Float16) * WPACK_ELEMS);
  _Float16* wpE   = (_Float16*)carve(sizeof(_Float16) * WPACK_ELEMS);
  float*    bpart = (float*)carve(sizeof(float) * CC * 64 * 2);
  float*    stats = (float*)carve(sizeof(float) * 64);
  float*    rpart = (float*)carve(sizeof(float) * BB * 64 * 4);

  // setup (per-launch, deterministic)
  hipMemsetAsync(pad, 0, padBytes, stream);
  hipMemcpyAsync(h, h0, sizeof(float) * NEL, hipMemcpyDeviceToDevice, stream);
  pack_w_kernel<<<(WPACK_ELEMS + 255) / 256, 256, 0, stream>>>(w_inh, wpI);
  pack_w_kernel<<<(WPACK_ELEMS + 255) / 256, 256, 0, stream>>>(w_exc, wpE);
  conv0_kernel<<<(NEL + 255) / 256, 256, 0, stream>>>(x, conv0_w, conv0_b, xin);

  const int convBlocks = (BB * YPAIRS * XTILES) / 4;   // 3000 blocks x 4 waves
  for (int t = 0; t < 8; ++t) {
    g1hg_kernel<<<(NPC + 255) / 256, 256, 0, stream>>>(h, u1_w, u1_b, pad);
    conv15_wmma_kernel<<<convBlocks, 128, 0, stream>>>(pad, wpI, craw);
    bnstats_kernel<<<dim3(64, CC), 256, 0, stream>>>(craw, bpart);
    bnfinal_kernel<<<1, 32, 0, stream>>>(bpart, stats);
    ns1_kernel<<<(NEL + 255) / 256, 256, 0, stream>>>(craw, stats, bn1_w, bn1_b,
                                                      xin, h, alpha, mu, pad);
    conv15_wmma_kernel<<<convBlocks, 128, 0, stream>>>(pad, wpE, craw);
    bnstats_kernel<<<dim3(64, CC), 256, 0, stream>>>(craw, bpart);
    bnfinal_kernel<<<1, 32, 0, stream>>>(bpart, stats);
    hupd_kernel<<<(NPC + 255) / 256, 256, 0, stream>>>(pad, craw, stats, bn3_w, bn3_b,
                                                       u2_w, u2_b, kappa, gam, w_mix, h);
  }

  readout_reduce_kernel<<<dim3(64, BB), 256, 0, stream>>>(h, conv6_w, conv6_b, rpart);
  readout_final_kernel<<<1, 64, 0, stream>>>(rpart, bno_w, bno_b, fc_w, fc_b,
                                             (float*)d_out);
}
